// HeteroGNN_67894843015626
// MI455X (gfx1250) — matrix operand order repaired and checked by reading, not compiled, not run on hardware
//
#include <hip/hip_runtime.h>
#include <stdint.h>

// ---------------------------------------------------------------------------
// HeteroGNN (2-layer hetero GraphSAGE) for MI455X / gfx1250.
//
// Math folding:
//   identity relations (pp, cc, uu) have mean == x, so per node type:
//     post_l    = x_post @ (Wl_pp + Wr_pp) + bl_pp
//     comment_l = mean_hc @ (0.5 Wl_hc) + x_c @ 0.5(Wr_hc + Wl_cc + Wr_cc) + 0.5(bl_hc+bl_cc)
//     user_l    = mean_ab @ (0.5 Wl_ab) + x_u @ 0.5(Wr_ab + Wl_uu + Wr_uu) + 0.5(bl_ab+bl_uu)
//   Only hc and ab need scatter-mean (float atomics, bandwidth bound).
//   Dense work: v_wmma_f32_16x16x32_bf16, double-buffered LDS pipeline.
//   B panels (already bf16) are moved with GLOBAL_LOAD_ASYNC_TO_LDS_B128
//   (ASYNCcnt) so the wave never touches B data; A is converted f32->bf16
//   in registers on the way to LDS (fused 1/deg scaling).
// ---------------------------------------------------------------------------

typedef __bf16 bf16;
typedef __attribute__((ext_vector_type(16))) __bf16 v16bf;
typedef __attribute__((ext_vector_type(8)))  float  v8f;

#define TILE_M   64
#define KSTEP    32
#define A_STRIDE 40   // bf16 elems per LDS row: 32 data + 8 pad (conflict-free)
#define B_STRIDE 40

// Low 32 bits of a generic pointer into the shared aperture are the
// wave-relative LDS byte offset (ISA 10.2).
__device__ __forceinline__ uint32_t lds_off(const void* p) {
  return (uint32_t)(uintptr_t)p;
}

// -------------------- small prep kernels --------------------

// out[n*K + k] = scale * (W0[k,n] + W1[k,n] + W2[k,n])   (transposed, bf16)
__global__ void combine_w_kernel(const float* __restrict__ W0,
                                 const float* __restrict__ W1,
                                 const float* __restrict__ W2,
                                 float scale, int K, bf16* __restrict__ outT) {
  int idx = blockIdx.x * blockDim.x + threadIdx.x;
  if (idx >= 256 * K) return;
  int n = idx / K, k = idx - n * K;
  float v = W0[k * 256 + n];
  if (W1) v += W1[k * 256 + n];
  if (W2) v += W2[k * 256 + n];
  outT[idx] = (bf16)(v * scale);
}

__global__ void combine_b_kernel(const float* __restrict__ b0,
                                 const float* __restrict__ b1,
                                 float scale, float* __restrict__ out) {
  int i = threadIdx.x;  // 256 threads
  float v = b0[i];
  if (b1) v += b1[i];
  out[i] = v * scale;
}

__global__ void count_dst_kernel(const int* __restrict__ dst, int E,
                                 float* __restrict__ cnt) {
  int e = blockIdx.x * blockDim.x + threadIdx.x;
  if (e < E) atomicAdd(&cnt[dst[e]], 1.0f);
}

// one wave per edge: agg[dst[e], :] += x[src[e], :]
__global__ void scatter_add_kernel(const float* __restrict__ x, int D,
                                   const int* __restrict__ src,
                                   const int* __restrict__ dst,
                                   int E, float* __restrict__ agg) {
  int e = blockIdx.x * 8 + (threadIdx.x >> 5);
  if (e >= E) return;
  int lane = threadIdx.x & 31;
  const float* xr = x + (size_t)src[e] * D;
  float*       ar = agg + (size_t)dst[e] * D;
  for (int i = lane; i < D; i += 32) atomicAdd(&ar[i], xr[i]);
}

// -------------------- fused SAGE GEMM (WMMA bf16, async B staging) ----------
//
// OUT[M x 256] = act( rowscale(A1) @ W1t^T + A2 @ W2t^T + bias )
//   A1: f32 [M x K1] (optional 1/max(cnt,1) row scaling), A2 optional.
//   W*t: bf16, N-major [256 x K] (row n = column n of the K x 256 weight).
//   Block: 256 thr = 8 waves; tile 64 x 256; wave = 16 x 128 (8 wmma accums).
//   Per K-step: A (64x32) converted f32->bf16 via registers into LDS,
//   B (256x32 bf16) copied global->LDS by the async engine (ASYNCcnt).

__launch_bounds__(256)
__global__ void sage_gemm_kernel(const float* __restrict__ A1,
                                 const float* __restrict__ cnt1, int K1,
                                 const float* __restrict__ A2, int K2,
                                 const bf16* __restrict__ W1t,
                                 const bf16* __restrict__ W2t,
                                 const float* __restrict__ bias,
                                 float* __restrict__ OUT, int M, int doRelu) {
  __shared__ __align__(16) bf16 sA[2][TILE_M * A_STRIDE];
  __shared__ __align__(16) bf16 sB[2][256 * B_STRIDE];

  const int tid  = threadIdx.x;
  const int lane = tid & 31;
  const int wid  = tid >> 5;
  const int wm   = wid >> 1;       // 0..3 : row group (16 rows each)
  const int wn   = wid & 1;        // 0..1 : col half (128 cols each)
  const int nlo  = lane & 15;
  const int khalf = lane >> 4;     // 0: lanes 0-15, 1: lanes 16-31
  const int colBase = wn * 128;
  const int blockRow = blockIdx.x * TILE_M;

  // C initialized with bias (all 8 rows of a lane share one column)
  v8f acc[8];
#pragma unroll
  for (int t = 0; t < 8; ++t) {
    float bv = bias[colBase + t * 16 + nlo];
    v8f a;
#pragma unroll
    for (int i = 0; i < 8; ++i) a[i] = bv;
    acc[t] = a;
  }

  // cooperative staging coords:
  //   A: each thread converts 8 f32 -> bf16 of row ldRow
  //   B: each thread owns one 64-byte row (col) of Wt, moved asynchronously
  const int ldRow = tid >> 2;
  const int ldCol = (tid & 3) * 8;
  const int gRow  = blockRow + ldRow;

  const uint32_t sBoff[2] = { lds_off(&sB[0][tid * B_STRIDE]),
                              lds_off(&sB[1][tid * B_STRIDE]) };

  int cur = 0;  // ping-pong buffer index, persists across phases

  for (int phase = 0; phase < 2; ++phase) {
    const float* A   = phase ? A2 : A1;
    if (!A) continue;
    const int    K   = phase ? K2 : K1;
    const bf16*  Wt  = phase ? W2t : W1t;
    const float* cnt = phase ? nullptr : cnt1;

    float rscale = 1.0f;
    if (cnt && gRow < M) rscale = 1.0f / fmaxf(cnt[gRow], 1.0f);

    const float* aRowPtr = A + (size_t)gRow * K;
    const bf16*  bRowPtr = Wt + (size_t)tid * K;

    // A staging registers (next K-slab)
    float4 af0, af1;

    auto loadRegsA = [&](int k0) {
      if (gRow < M) {
        const float* sp = aRowPtr + k0 + ldCol;
        af0 = ((const float4*)sp)[0];
        af1 = ((const float4*)sp)[1];
        if (k0 + KSTEP < K)  // gfx1250 global_prefetch_b8, one slab ahead
          __builtin_prefetch(sp + KSTEP, 0, 0);
      } else {
        af0 = float4{0.f, 0.f, 0.f, 0.f};
        af1 = float4{0.f, 0.f, 0.f, 0.f};
      }
    };

    // async copy of this thread's 64-byte B row slab into sB[buf]
    auto issueAsyncB = [&](int k0, int buf) {
      uint64_t ga = (uint64_t)(uintptr_t)(bRowPtr + k0);
      uint32_t lo = sBoff[buf];
      asm volatile(
          "global_load_async_to_lds_b128 %0, %1, off\n\t"
          "global_load_async_to_lds_b128 %0, %1, off offset:16"
          :: "v"(lo), "v"(ga) : "memory");
    };

    auto storeLDS_A = [&](int buf) {
      union { bf16 h[8]; uint4 u; } pk;
      pk.h[0] = (bf16)(af0.x * rscale); pk.h[1] = (bf16)(af0.y * rscale);
      pk.h[2] = (bf16)(af0.z * rscale); pk.h[3] = (bf16)(af0.w * rscale);
      pk.h[4] = (bf16)(af1.x * rscale); pk.h[5] = (bf16)(af1.y * rscale);
      pk.h[6] = (bf16)(af1.z * rscale); pk.h[7] = (bf16)(af1.w * rscale);
      *(uint4*)&sA[buf][ldRow * A_STRIDE + ldCol] = pk.u;
    };

    issueAsyncB(0, cur);   // prologue: B(0) in flight while A(0) loads
    loadRegsA(0);

    for (int k0 = 0; k0 < K; k0 += KSTEP) {
      storeLDS_A(cur);
      // our async B rows for this buffer must be resident before the barrier
      asm volatile("s_wait_asynccnt 0x0" ::: "memory");
      __syncthreads();
      if (k0 + KSTEP < K) {      // overlap next slab with compute below
        issueAsyncB(k0 + KSTEP, cur ^ 1);
        loadRegsA(k0 + KSTEP);
      }

      // A operand: 16x32 bf16, CDNA5 layout:
      //   lanes 0-15 : row=lane,    elems 0-7 = K 0-7,  elems 8-15 = K 16-23
      //   lanes 16-31: row=lane-16, elems 0-7 = K 8-15, elems 8-15 = K 24-31
      const int aOff = (wm * 16 + nlo) * A_STRIDE + (khalf << 3);
      union { v16bf v; uint4 u[2]; } aop;
      aop.u[0] = *(const uint4*)&sA[cur][aOff];
      aop.u[1] = *(const uint4*)&sA[cur][aOff + 16];

      // B operand: 32x16 bf16; same K layout as A: lanes 0-15 K 0-7,16-23;
      // lanes 16-31 K 8-15,24-31 of column n in the staged sB row.
#pragma unroll
      for (int t = 0; t < 8; ++t) {
        const int bOff = (colBase + t * 16 + nlo) * B_STRIDE + (khalf << 3);
        union { v16bf v; uint4 u[2]; } bop;
        bop.u[0] = *(const uint4*)&sB[cur][bOff];
        bop.u[1] = *(const uint4*)&sB[cur][bOff + 16];
        acc[t] = __builtin_amdgcn_wmma_f32_16x16x32_bf16(
            false, aop.v, false, bop.v, (short)0, acc[t], false, false);
      }
      cur ^= 1;
    }
  }

  // D layout: elem v of lane -> row = v + 8*khalf, col = nlo (per 16x16 tile)
  const int rBase = blockRow + wm * 16 + (khalf << 3);
#pragma unroll
  for (int t = 0; t < 8; ++t) {
    const int c = colBase + t * 16 + nlo;
#pragma unroll
    for (int v = 0; v < 8; ++v) {
      const int r = rBase + v;
      if (r < M) {
        float val = acc[t][v];
        if (doRelu) val = fmaxf(val, 0.0f);
        OUT[(size_t)r * 256 + c] = val;
      }
    }
  }
}

// -------------------- host orchestration --------------------

extern "C" void kernel_launch(void* const* d_in, const int* in_sizes, int n_in,
                              void* d_out, int out_size, void* d_ws, size_t ws_size,
                              hipStream_t stream) {
  (void)n_in; (void)out_size; (void)ws_size;

  const float* x_post    = (const float*)d_in[0];
  const float* x_comment = (const float*)d_in[1];
  const float* x_user    = (const float*)d_in[2];
  const int*   ei_hc     = (const int*)d_in[3];
  const int*   ei_ab     = (const int*)d_in[4];
  // d_in[5..7] = ei_pp / ei_cc / ei_uu: identity edges, folded analytically.

  const int NP = in_sizes[0] / 256;   // 100000
  const int NC = in_sizes[1] / 128;   // 500000
  const int NU = in_sizes[2] / 64;    // 200000
  const int EHC = in_sizes[3] / 2;    // 500000
  const int EAB = in_sizes[4] / 2;    // 500000

  // params assumed in setup_inputs() insertion order:
  //   for layer in (1,2): for rel in (hc, ab, pp, cc, uu): Wl, bl, Wr
  const float* P[30];
  for (int i = 0; i < 30; ++i) P[i] = (const float*)d_in[8 + i];
  enum { HC1 = 0, AB1 = 3, PP1 = 6, CC1 = 9, UU1 = 12,
         HC2 = 15, AB2 = 18, PP2 = 21, CC2 = 24, UU2 = 27 };
#define WLP(r) P[(r) + 0]
#define BLP(r) P[(r) + 1]
#define WRP(r) P[(r) + 2]

  // ---- workspace carve ----
  char* ws = (char*)d_ws;
  size_t off = 0;
  auto carve = [&](size_t bytes) -> void* {
    void* p = ws + off;
    off = (off + bytes + 255) & ~(size_t)255;
    return p;
  };
  bf16* w_pp1 = (bf16*)carve(256 * 256 * sizeof(bf16));
  bf16* w_hc1 = (bf16*)carve(256 * 256 * sizeof(bf16));
  bf16* w_c1  = (bf16*)carve(256 * 128 * sizeof(bf16));
  bf16* w_ab1 = (bf16*)carve(256 * 128 * sizeof(bf16));
  bf16* w_u1  = (bf16*)carve(256 * 64  * sizeof(bf16));
  bf16* w_pp2 = (bf16*)carve(256 * 256 * sizeof(bf16));
  bf16* w_hc2 = (bf16*)carve(256 * 256 * sizeof(bf16));
  bf16* w_c2  = (bf16*)carve(256 * 256 * sizeof(bf16));
  bf16* w_ab2 = (bf16*)carve(256 * 256 * sizeof(bf16));
  bf16* w_u2  = (bf16*)carve(256 * 256 * sizeof(bf16));
  float* b_p1 = (float*)carve(256 * 4);
  float* b_c1 = (float*)carve(256 * 4);
  float* b_u1 = (float*)carve(256 * 4);
  float* b_p2 = (float*)carve(256 * 4);
  float* b_c2 = (float*)carve(256 * 4);
  float* b_u2 = (float*)carve(256 * 4);
  float* cnt_c = (float*)carve((size_t)NC * 4);
  float* cnt_u = (float*)carve((size_t)NU * 4);
  float* agg_c = (float*)carve((size_t)NC * 256 * 4);  // hc agg, L1 then L2
  float* agg_u = (float*)carve((size_t)NU * 256 * 4);  // ab agg, L1 then L2
  float* h_post = (float*)carve((size_t)NP * 256 * 4);
  float* h_comm = (float*)carve((size_t)NC * 256 * 4);
  float* h_user = (float*)carve((size_t)NU * 256 * 4);

  float* out_post = (float*)d_out;
  float* out_comm = out_post + (size_t)NP * 256;
  float* out_user = out_comm + (size_t)NC * 256;

  // ---- 1. fold weights/biases (tiny) ----
  auto cw = [&](const float* W0, const float* W1, const float* W2,
                float s, int K, bf16* out) {
    hipLaunchKernelGGL(combine_w_kernel, dim3((256 * K + 255) / 256), dim3(256),
                       0, stream, W0, W1, W2, s, K, out);
  };
  auto cb = [&](const float* b0, const float* b1, float s, float* out) {
    hipLaunchKernelGGL(combine_b_kernel, dim3(1), dim3(256), 0, stream, b0, b1, s, out);
  };
  cw(WLP(PP1), WRP(PP1), nullptr, 1.0f, 256, w_pp1);
  cw(WLP(HC1), nullptr,  nullptr, 0.5f, 256, w_hc1);
  cw(WRP(HC1), WLP(CC1), WRP(CC1), 0.5f, 128, w_c1);
  cw(WLP(AB1), nullptr,  nullptr, 0.5f, 128, w_ab1);
  cw(WRP(AB1), WLP(UU1), WRP(UU1), 0.5f, 64, w_u1);
  cw(WLP(PP2), WRP(PP2), nullptr, 1.0f, 256, w_pp2);
  cw(WLP(HC2), nullptr,  nullptr, 0.5f, 256, w_hc2);
  cw(WRP(HC2), WLP(CC2), WRP(CC2), 0.5f, 256, w_c2);
  cw(WLP(AB2), nullptr,  nullptr, 0.5f, 256, w_ab2);
  cw(WRP(AB2), WLP(UU2), WRP(UU2), 0.5f, 256, w_u2);
  cb(BLP(PP1), nullptr,  1.0f, b_p1);
  cb(BLP(HC1), BLP(CC1), 0.5f, b_c1);
  cb(BLP(AB1), BLP(UU1), 0.5f, b_u1);
  cb(BLP(PP2), nullptr,  1.0f, b_p2);
  cb(BLP(HC2), BLP(CC2), 0.5f, b_c2);
  cb(BLP(AB2), BLP(UU2), 0.5f, b_u2);

  // ---- 2. degree counts (shared across layers) ----
  hipMemsetAsync(cnt_c, 0, (size_t)NC * 4, stream);
  hipMemsetAsync(cnt_u, 0, (size_t)NU * 4, stream);
  hipLaunchKernelGGL(count_dst_kernel, dim3((EHC + 255) / 256), dim3(256), 0,
                     stream, ei_hc + EHC, EHC, cnt_c);
  hipLaunchKernelGGL(count_dst_kernel, dim3((EAB + 255) / 256), dim3(256), 0,
                     stream, ei_ab + EAB, EAB, cnt_u);

  // ---- 3. layer-1 aggregations ----
  hipMemsetAsync(agg_c, 0, (size_t)NC * 256 * 4, stream);
  hipMemsetAsync(agg_u, 0, (size_t)NU * 128 * 4, stream);
  hipLaunchKernelGGL(scatter_add_kernel, dim3((EHC + 7) / 8), dim3(256), 0,
                     stream, x_post, 256, ei_hc, ei_hc + EHC, EHC, agg_c);
  hipLaunchKernelGGL(scatter_add_kernel, dim3((EAB + 7) / 8), dim3(256), 0,
                     stream, x_comment, 128, ei_ab, ei_ab + EAB, EAB, agg_u);

  auto gemm = [&](const float* A1, const float* cnt, int K1,
                  const float* A2, int K2, const bf16* W1t, const bf16* W2t,
                  const float* bias, float* out, int M, int relu) {
    hipLaunchKernelGGL(sage_gemm_kernel, dim3((M + TILE_M - 1) / TILE_M),
                       dim3(256), 0, stream,
                       A1, cnt, K1, A2, K2, W1t, W2t, bias, out, M, relu);
  };

  // ---- 4. layer-1 fused GEMMs (+ReLU) ----
  gemm(x_post, nullptr, 256, nullptr, 0,   w_pp1, nullptr, b_p1, h_post, NP, 1);
  gemm(agg_c,  cnt_c,   256, x_comment, 128, w_hc1, w_c1,  b_c1, h_comm, NC, 1);
  gemm(agg_u,  cnt_u,   128, x_user,    64,  w_ab1, w_u1,  b_u1, h_user, NU, 1);

  // ---- 5. layer-2 aggregations (reuse agg buffers) ----
  hipMemsetAsync(agg_c, 0, (size_t)NC * 256 * 4, stream);
  hipMemsetAsync(agg_u, 0, (size_t)NU * 256 * 4, stream);
  hipLaunchKernelGGL(scatter_add_kernel, dim3((EHC + 7) / 8), dim3(256), 0,
                     stream, h_post, 256, ei_hc, ei_hc + EHC, EHC, agg_c);
  hipLaunchKernelGGL(scatter_add_kernel, dim3((EAB + 7) / 8), dim3(256), 0,
                     stream, h_comm, 256, ei_ab, ei_ab + EAB, EAB, agg_u);

  // ---- 6. layer-2 fused GEMMs (no ReLU) -> d_out ----
  gemm(h_post, nullptr, 256, nullptr, 0,   w_pp2, nullptr, b_p2, out_post, NP, 0);
  gemm(agg_c,  cnt_c,   256, h_comm, 256,  w_hc2, w_c2,   b_c2, out_comm, NC, 0);
  gemm(agg_u,  cnt_u,   256, h_user, 256,  w_ab2, w_u2,   b_u2, out_user, NU, 0);
}
